// SurrogateSNN_36352603193486
// MI455X (gfx1250) — matrix operand order repaired and checked by reading, not compiled, not run on hardware
//
#include <hip/hip_runtime.h>
#include <hip/hip_bf16.h>

typedef _Float16 h16v __attribute__((ext_vector_type(16)));
typedef _Float16 h8v  __attribute__((ext_vector_type(8)));
typedef float    f8v  __attribute__((ext_vector_type(8)));

#define SNN_B    128
#define SNN_THR  1.0f
#define SNN_BETA 0.9f

// ---------------------------------------------------------------------------
// Bernoulli rate-encoded spike input for one time step.
// x: [128][3][32][32] NCHW f32.  out: [128][32][32][3] NHWC f16 (for im2col).
// ---------------------------------------------------------------------------
__global__ __launch_bounds__(256) void snn_spikegen(const float* __restrict__ x,
                                                    _Float16* __restrict__ out,
                                                    int t) {
  int i = blockIdx.x * blockDim.x + threadIdx.x;
  const int total = SNN_B * 32 * 32 * 3;
  if (i >= total) return;
  int c = i % 3;  int r = i / 3;
  int xx = r % 32; r /= 32;
  int y  = r % 32; int b = r / 32;
  float p = x[((b * 3 + c) * 32 + y) * 32 + xx];
  p = fminf(fmaxf((p + 1.0f) * 0.5f, 0.0f), 1.0f);
  // Deterministic per-(t, element) hash RNG (Wellons triple-xorshift-mult).
  unsigned s = (unsigned)i + 0x9E3779B9u * (unsigned)(t + 1);
  s ^= s >> 16; s *= 0x7FEB352Du;
  s ^= s >> 15; s *= 0x846CA68Bu;
  s ^= s >> 16;
  float u = (float)(s >> 8) * (1.0f / 16777216.0f);
  out[i] = (_Float16)((u < p) ? 1.0f : 0.0f);
}

// ---------------------------------------------------------------------------
// im2col for 3x3 SAME conv, NHWC input, K = C*9 padded to Kpad (zeros).
// in: [128][H][W][C] f16.  col: [128*H*W][Kpad] f16, k = cin*9 + ky*3 + kx.
// ---------------------------------------------------------------------------
__global__ __launch_bounds__(256) void snn_im2col(const _Float16* __restrict__ in,
                                                  _Float16* __restrict__ col,
                                                  int H, int W, int C, int Kpad) {
  int K = C * 9;
  size_t total = (size_t)SNN_B * H * W * Kpad;
  size_t i = (size_t)blockIdx.x * blockDim.x + threadIdx.x;
  if (i >= total) return;
  int k = (int)(i % (size_t)Kpad);
  size_t m = i / (size_t)Kpad;
  _Float16 v = (_Float16)0.0f;
  if (k < K) {
    int xo = (int)(m % (size_t)W); size_t r = m / (size_t)W;
    int yo = (int)(r % (size_t)H); int b = (int)(r / (size_t)H);
    int cin = k / 9; int rr = k % 9; int ky = rr / 3; int kx = rr % 3;
    int yy = yo + ky - 1, xx = xo + kx - 1;
    if (yy >= 0 && yy < H && xx >= 0 && xx < W)
      v = in[(((size_t)b * H + yy) * W + xx) * C + cin];
  }
  col[i] = v;
}

// ---------------------------------------------------------------------------
// fp32 -> f16 weight conversion with row (N) and column (K) zero padding.
// src: [Nsrc][K] f32.  dst: [Npad][Kpad] f16.
// ---------------------------------------------------------------------------
__global__ __launch_bounds__(256) void snn_cvt_pad(const float* __restrict__ src,
                                                   _Float16* __restrict__ dst,
                                                   int Nsrc, int K, int Npad, int Kpad) {
  size_t total = (size_t)Npad * Kpad;
  size_t i = (size_t)blockIdx.x * blockDim.x + threadIdx.x;
  if (i >= total) return;
  int k = (int)(i % (size_t)Kpad);
  int n = (int)(i / (size_t)Kpad);
  float v = (n < Nsrc && k < K) ? src[(size_t)n * K + k] : 0.0f;
  dst[i] = (_Float16)v;
}

// ---------------------------------------------------------------------------
// 2x2 maxpool on NHWC f16 spikes. nchw_flat=1 writes [b][c*Ho*Ho + yo*Ho + xo]
// (NCHW flatten, needed before fc1 to match the reference reshape).
// ---------------------------------------------------------------------------
__global__ __launch_bounds__(256) void snn_pool2(const _Float16* __restrict__ in,
                                                 _Float16* __restrict__ out,
                                                 int Hin, int C, int nchw_flat) {
  int Ho = Hin >> 1;
  size_t total = (size_t)SNN_B * Ho * Ho * C;
  size_t i = (size_t)blockIdx.x * blockDim.x + threadIdx.x;
  if (i >= total) return;
  int c  = (int)(i % (size_t)C); size_t r = i / (size_t)C;
  int xo = (int)(r % (size_t)Ho); r /= (size_t)Ho;
  int yo = (int)(r % (size_t)Ho);
  int b  = (int)(r / (size_t)Ho);
  const _Float16* p = in + (((size_t)b * Hin + 2 * yo) * Hin + 2 * xo) * C + c;
  float m0 = fmaxf((float)p[0], (float)p[C]);
  float m1 = fmaxf((float)p[(size_t)Hin * C], (float)p[(size_t)Hin * C + C]);
  float mx = fmaxf(m0, m1);
  size_t oidx = nchw_flat
      ? ((size_t)b * C + c) * (Ho * Ho) + (size_t)yo * Ho + xo
      : i;
  out[oidx] = (_Float16)mx;
}

// ---------------------------------------------------------------------------
// Fused GEMM + LIF with MTxNT register blocking (MT*NT WMMAs per K-step).
//   cur[m][n] = sum_k A[m][k] * Bt[n][k] + bias[n]       (WMMA f32<-f16)
//   reset = (mem_old > 1);  mem = 0.9*mem_old + cur - reset;  spk = (mem > 1)
// A:   [M][K]    f16, K % 32 == 0, rows 16B-aligned
// Bt:  [Npad][K] f16 (natural [out][in...] weight layout)
// mem: [M][Npad] f32 (persistent state),  spk: [M][Npad] f16
// rec_spk/rec_mem: optional [M][Nvalid] f32 slices of d_out (fc3 only).
// ---------------------------------------------------------------------------
template <int MT, int NT>
__global__ __launch_bounds__(256) void snn_gemm_lif(
    const _Float16* __restrict__ A,
    const _Float16* __restrict__ Bt,
    const float* __restrict__ bias,
    float* __restrict__ mem,
    _Float16* __restrict__ spk,
    float* __restrict__ rec_spk,
    float* __restrict__ rec_mem,
    int M, int Npad, int Nvalid, int K) {
  int tid  = blockIdx.x * blockDim.x + threadIdx.x;
  int wid  = tid >> 5;
  int lane = threadIdx.x & 31;
  int ntn  = Npad / (16 * NT);
  int nTiles = (M / (16 * MT)) * ntn;
  if (wid >= nTiles) return;  // uniform per wave: EXEC stays all-ones for WMMA

  int mt = wid / ntn;
  int nt = wid - mt * ntn;
  int n0    = nt * (16 * NT) + (lane & 15);
  int aRow0 = mt * (16 * MT) + (lane & 15);
  // ISA 16-bit A layout: lanes<16 hold K in [kb..kb+7]+[kb+16..kb+23],
  // lanes>=16 hold [kb+8..kb+15]+[kb+24..kb+31].
  int aOff = (lane & 16) ? 8 : 0;
  // 16-bit B layout (Bt[n][k] row-major), same K order as A: lanes<16 hold K
  // [kb..kb+7]+[kb+16..kb+23], lanes>=16 hold [kb+8..kb+15]+[kb+24..kb+31].
  int bOff = (lane & 16) ? 8 : 0;

  const _Float16* aBase[MT];
#pragma unroll
  for (int mi = 0; mi < MT; ++mi)
    aBase[mi] = A + (size_t)(aRow0 + 16 * mi) * K + aOff;
  const _Float16* bBase[NT];
#pragma unroll
  for (int ni = 0; ni < NT; ++ni)
    bBase[ni] = Bt + (size_t)(n0 + 16 * ni) * K + bOff;

  f8v c[MT][NT];
#pragma unroll
  for (int mi = 0; mi < MT; ++mi)
#pragma unroll
    for (int ni = 0; ni < NT; ++ni) c[mi][ni] = {};

  for (int kb = 0; kb < K; kb += 32) {
    h16v av[MT], bv[NT];
#pragma unroll
    for (int mi = 0; mi < MT; ++mi) {
      h8v a0 = *(const h8v*)(aBase[mi] + kb);
      h8v a1 = *(const h8v*)(aBase[mi] + kb + 16);
      av[mi] = __builtin_shufflevector(a0, a1, 0, 1, 2, 3, 4, 5, 6, 7,
                                       8, 9, 10, 11, 12, 13, 14, 15);
    }
#pragma unroll
    for (int ni = 0; ni < NT; ++ni) {
      h8v b0 = *(const h8v*)(bBase[ni] + kb);
      h8v b1 = *(const h8v*)(bBase[ni] + kb + 16);
      bv[ni] = __builtin_shufflevector(b0, b1, 0, 1, 2, 3, 4, 5, 6, 7,
                                       8, 9, 10, 11, 12, 13, 14, 15);
    }
#pragma unroll
    for (int mi = 0; mi < MT; ++mi)
#pragma unroll
      for (int ni = 0; ni < NT; ++ni)
        c[mi][ni] = __builtin_amdgcn_wmma_f32_16x16x32_f16(
            false, av[mi], false, bv[ni], (short)0, c[mi][ni], false, false);
  }

  // C/D layout: VGPR v holds M = tile_m + v (+8 for lanes>=16), N = lane&15.
#pragma unroll
  for (int ni = 0; ni < NT; ++ni) {
    int n = n0 + 16 * ni;
    float bn = (n < Nvalid) ? bias[n] : 0.0f;
#pragma unroll
    for (int mi = 0; mi < MT; ++mi) {
      int mBase = mt * (16 * MT) + 16 * mi + ((lane & 16) ? 8 : 0);
#pragma unroll
      for (int v = 0; v < 8; ++v) {
        int m = mBase + v;
        size_t idx = (size_t)m * Npad + n;
        float cur   = c[mi][ni][v] + bn;
        float mo    = mem[idx];
        float reset = (mo > SNN_THR) ? 1.0f : 0.0f;
        float mn    = SNN_BETA * mo + cur - reset * SNN_THR;
        mem[idx] = mn;
        float s = (mn - SNN_THR > 0.0f) ? 1.0f : 0.0f;
        spk[idx] = (_Float16)s;
        if (rec_spk != nullptr && n < Nvalid) {
          rec_spk[(size_t)m * Nvalid + n] = s;
          rec_mem[(size_t)m * Nvalid + n] = mn;
        }
      }
    }
  }
}

// ---------------------------------------------------------------------------
// Host-side orchestration
// ---------------------------------------------------------------------------
static inline int grid1d(size_t total) { return (int)((total + 255) / 256); }

template <int MT, int NT>
static void launch_gemm_lif(hipStream_t s, const _Float16* A, const _Float16* Bt,
                            const float* bias, float* mem, _Float16* spk,
                            float* rspk, float* rmem,
                            int M, int Npad, int Nvalid, int K) {
  int waves = (M / (16 * MT)) * (Npad / (16 * NT));
  int blocks = (waves * 32 + 255) / 256;
  snn_gemm_lif<MT, NT><<<blocks, 256, 0, s>>>(A, Bt, bias, mem, spk, rspk, rmem,
                                              M, Npad, Nvalid, K);
}

extern "C" void kernel_launch(void* const* d_in, const int* in_sizes, int n_in,
                              void* d_out, int out_size, void* d_ws, size_t ws_size,
                              hipStream_t stream) {
  (void)in_sizes; (void)n_in; (void)out_size; (void)ws_size;
  const float* x   = (const float*)d_in[0];
  const float* w1  = (const float*)d_in[1];  const float* b1  = (const float*)d_in[2];
  const float* w2  = (const float*)d_in[3];  const float* b2  = (const float*)d_in[4];
  const float* w3  = (const float*)d_in[5];  const float* b3  = (const float*)d_in[6];
  const float* fw1 = (const float*)d_in[7];  const float* fb1 = (const float*)d_in[8];
  const float* fw2 = (const float*)d_in[9];  const float* fb2 = (const float*)d_in[10];
  const float* fw3 = (const float*)d_in[11]; const float* fb3 = (const float*)d_in[12];
  const int T = 25;  // num_steps (device scalar; fixed by setup_inputs)

  float* out_spk = (float*)d_out;                    // [25][128][10]
  float* out_mem = (float*)d_out + T * SNN_B * 10;   // [25][128][10]

  // Bump allocator over d_ws, 256B aligned.
  size_t off = 0;
  auto alloc = [&](size_t bytes) -> char* {
    char* p = (char*)d_ws + off;
    off = (off + bytes + 255) & ~(size_t)255;
    return p;
  };
  _Float16* spk_in = (_Float16*)alloc((size_t)SNN_B * 32 * 32 * 3 * 2);
  _Float16* col    = (_Float16*)alloc((size_t)32768 * 288 * 2);     // max im2col (conv2)
  float*    mem1 = (float*)alloc((size_t)131072 * 32 * 4);
  _Float16* spk1 = (_Float16*)alloc((size_t)131072 * 32 * 2);
  _Float16* x1   = (_Float16*)alloc((size_t)SNN_B * 16 * 16 * 32 * 2);
  float*    mem2 = (float*)alloc((size_t)32768 * 64 * 4);
  _Float16* spk2 = (_Float16*)alloc((size_t)32768 * 64 * 2);
  _Float16* x2   = (_Float16*)alloc((size_t)SNN_B * 8 * 8 * 64 * 2);
  float*    mem3 = (float*)alloc((size_t)8192 * 128 * 4);
  _Float16* spk3 = (_Float16*)alloc((size_t)8192 * 128 * 2);
  _Float16* x3   = (_Float16*)alloc((size_t)SNN_B * 2048 * 2);      // NCHW-flat
  float*    mem4 = (float*)alloc((size_t)SNN_B * 256 * 4);
  _Float16* spk4 = (_Float16*)alloc((size_t)SNN_B * 256 * 2);
  float*    mem5 = (float*)alloc((size_t)SNN_B * 128 * 4);
  _Float16* spk5 = (_Float16*)alloc((size_t)SNN_B * 128 * 2);
  float*    mem6 = (float*)alloc((size_t)SNN_B * 16 * 4);
  _Float16* spk6 = (_Float16*)alloc((size_t)SNN_B * 16 * 2);
  _Float16* w1h  = (_Float16*)alloc((size_t)32 * 32 * 2);           // K 27 -> 32
  _Float16* w2h  = (_Float16*)alloc((size_t)64 * 288 * 2);
  _Float16* w3h  = (_Float16*)alloc((size_t)128 * 576 * 2);
  _Float16* fw1h = (_Float16*)alloc((size_t)256 * 2048 * 2);
  _Float16* fw2h = (_Float16*)alloc((size_t)128 * 256 * 2);
  _Float16* fw3h = (_Float16*)alloc((size_t)16 * 128 * 2);          // N 10 -> 16

  // Deterministic state init (harness does not re-poison between replays).
  hipMemsetAsync(mem1, 0, (size_t)131072 * 32 * 4, stream);
  hipMemsetAsync(mem2, 0, (size_t)32768 * 64 * 4, stream);
  hipMemsetAsync(mem3, 0, (size_t)8192 * 128 * 4, stream);
  hipMemsetAsync(mem4, 0, (size_t)SNN_B * 256 * 4, stream);
  hipMemsetAsync(mem5, 0, (size_t)SNN_B * 128 * 4, stream);
  hipMemsetAsync(mem6, 0, (size_t)SNN_B * 16 * 4, stream);

  // Weight conversion (f32 -> f16, padded) — cheap, once per call.
  snn_cvt_pad<<<grid1d((size_t)32 * 32), 256, 0, stream>>>(w1, w1h, 32, 27, 32, 32);
  snn_cvt_pad<<<grid1d((size_t)64 * 288), 256, 0, stream>>>(w2, w2h, 64, 288, 64, 288);
  snn_cvt_pad<<<grid1d((size_t)128 * 576), 256, 0, stream>>>(w3, w3h, 128, 576, 128, 576);
  snn_cvt_pad<<<grid1d((size_t)256 * 2048), 256, 0, stream>>>(fw1, fw1h, 256, 2048, 256, 2048);
  snn_cvt_pad<<<grid1d((size_t)128 * 256), 256, 0, stream>>>(fw2, fw2h, 128, 256, 128, 256);
  snn_cvt_pad<<<grid1d((size_t)16 * 128), 256, 0, stream>>>(fw3, fw3h, 10, 128, 16, 128);

  for (int t = 0; t < T; ++t) {
    // Input spike train for this step (NHWC f16).
    snn_spikegen<<<grid1d((size_t)SNN_B * 32 * 32 * 3), 256, 0, stream>>>(x, spk_in, t);

    // conv1 (3->32, 32x32) + LIF, then pool -> 16x16
    snn_im2col<<<grid1d((size_t)131072 * 32), 256, 0, stream>>>(spk_in, col, 32, 32, 3, 32);
    launch_gemm_lif<2, 2>(stream, col, w1h, b1, mem1, spk1, nullptr, nullptr,
                          131072, 32, 32, 32);
    snn_pool2<<<grid1d((size_t)SNN_B * 16 * 16 * 32), 256, 0, stream>>>(spk1, x1, 32, 32, 0);

    // conv2 (32->64, 16x16) + LIF, then pool -> 8x8
    snn_im2col<<<grid1d((size_t)32768 * 288), 256, 0, stream>>>(x1, col, 16, 16, 32, 288);
    launch_gemm_lif<2, 2>(stream, col, w2h, b2, mem2, spk2, nullptr, nullptr,
                          32768, 64, 64, 288);
    snn_pool2<<<grid1d((size_t)SNN_B * 8 * 8 * 64), 256, 0, stream>>>(spk2, x2, 16, 64, 0);

    // conv3 (64->128, 8x8) + LIF, then pool -> 4x4, NCHW-flattened [128][2048]
    snn_im2col<<<grid1d((size_t)8192 * 576), 256, 0, stream>>>(x2, col, 8, 8, 64, 576);
    launch_gemm_lif<2, 2>(stream, col, w3h, b3, mem3, spk3, nullptr, nullptr,
                          8192, 128, 128, 576);
    snn_pool2<<<grid1d((size_t)SNN_B * 4 * 4 * 128), 256, 0, stream>>>(spk3, x3, 8, 128, 1);

    // fc1 (2048->256), fc2 (256->128), fc3 (128->10, padded to 16; records out)
    launch_gemm_lif<2, 2>(stream, x3, fw1h, fb1, mem4, spk4, nullptr, nullptr,
                          SNN_B, 256, 256, 2048);
    launch_gemm_lif<2, 2>(stream, spk4, fw2h, fb2, mem5, spk5, nullptr, nullptr,
                          SNN_B, 128, 128, 256);
    launch_gemm_lif<2, 1>(stream, spk5, fw3h, fb3, mem6, spk6,
                          out_spk + (size_t)t * SNN_B * 10,
                          out_mem + (size_t)t * SNN_B * 10,
                          SNN_B, 16, 10, 128);
  }
}
